// P2M_Attention_72121090834685
// MI455X (gfx1250) — compile-verified, matrix operand order checked by reading, not run on hardware
//
#include <hip/hip_runtime.h>
#include <stdint.h>

// ---------------------------------------------------------------------------
// CDNA5 (gfx1250) wave32 WMMA implementation of the P2M attention block.
// All GEMM-shaped math goes through v_wmma_f32_16x16x32_bf16 (bf16 in, fp32
// accum). Attention is flash-style so the [8,8,1024,2267] score tensor never
// hits HBM. Weights are pre-transposed to [N,K] bf16 so both A and B WMMA
// fragments load as two contiguous 16-byte chunks per lane (ISA 7.12.2
// 16-bit operand layout: lane<16 holds K {0..7,16..23}, lane>=16 holds
// K {8..15,24..31}).
//
// Round-2 changes:
//  * gemm_strip4 is software-pipelined (double-buffered fragments): next
//    K-step loads are issued before the current 4 WMMAs, so waits don't
//    drain loadcnt to 0 in front of every WMMA
//  * flash attention loads all 4 K-fragments as one cluster, and issues the
//    4 V-fragments early so the softmax VALU/LDS phase hides their latency
// ---------------------------------------------------------------------------

typedef __attribute__((ext_vector_type(16))) __bf16          v16bf;
typedef __attribute__((ext_vector_type(8)))  float           v8f;
typedef __attribute__((ext_vector_type(8)))  unsigned short  ushort8;

#define BATCH  8
#define CCH    512
#define LTOK   1024
#define NTOK   1243      // 21^2 + 16^2 + 13^2 + 11^2 + 16^2
#define NPAD   2272      // (1243 + 1024) padded to multiple of 32
#define NREAL  2267
#define MXOFF  987       // offset of max-pool tokens inside the 1243

__device__ __forceinline__ unsigned short f2bf(float f) {
  unsigned int u = __builtin_bit_cast(unsigned int, f);
  u += 0x7fffu + ((u >> 16) & 1u);                 // round-to-nearest-even
  return (unsigned short)(u >> 16);
}

union FragU { v16bf bf; ushort8 h[2]; unsigned short u[16]; };

// Load one 16x32 (bf16) A/B^T fragment. `row` is the matrix row this lane
// owns (M for A, N for B^T); `kbase` must be a multiple of 32.
__device__ __forceinline__ v16bf load_frag(const unsigned short* __restrict__ base,
                                           int ld, int row, int kbase, int lane) {
  FragU f;
  const unsigned short* p = base + (size_t)row * ld + kbase + ((lane >> 4) & 1) * 8;
  f.h[0] = *(const ushort8*)(p);        // K = khalf*8 + 0..7
  f.h[1] = *(const ushort8*)(p + 16);   // K = 16 + khalf*8 + 0..7
  return f.bf;
}

__device__ __forceinline__ v8f wmma_bf16(v16bf a, v16bf b, v8f c) {
  return __builtin_amdgcn_wmma_f32_16x16x32_bf16(false, a, false, b, (short)0, c,
                                                 false, false);
}

// 16x64 output strip: rows [r0,r0+16) (clamped to rmax-1; garbage rows are
// discarded at store), cols [c0,c0+64). Bt is [Ncols x K] bf16.
// Software-pipelined: next K-step fragments are loaded before current WMMAs.
__device__ __forceinline__ void gemm_strip4(const unsigned short* __restrict__ A, int lda,
                                            int r0, int rmax,
                                            const unsigned short* __restrict__ Bt, int ldb,
                                            int c0, int K, int lane, v8f acc[4]) {
  int rr = r0 + (lane & 15);
  rr = rr < rmax ? rr : rmax - 1;                  // clamp: branch-free edge tiles
  int cc = c0 + (lane & 15);
  v8f zz = {0.f, 0.f, 0.f, 0.f, 0.f, 0.f, 0.f, 0.f};
  for (int t = 0; t < 4; ++t) acc[t] = zz;

  v16bf a_cur = load_frag(A, lda, rr, 0, lane);
  v16bf b_cur0 = load_frag(Bt, ldb, cc,      0, lane);
  v16bf b_cur1 = load_frag(Bt, ldb, cc + 16, 0, lane);
  v16bf b_cur2 = load_frag(Bt, ldb, cc + 32, 0, lane);
  v16bf b_cur3 = load_frag(Bt, ldb, cc + 48, 0, lane);

  for (int k = 32; k < K; k += 32) {
    v16bf a_nxt  = load_frag(A, lda, rr, k, lane);
    v16bf b_nxt0 = load_frag(Bt, ldb, cc,      k, lane);
    v16bf b_nxt1 = load_frag(Bt, ldb, cc + 16, k, lane);
    v16bf b_nxt2 = load_frag(Bt, ldb, cc + 32, k, lane);
    v16bf b_nxt3 = load_frag(Bt, ldb, cc + 48, k, lane);
    acc[0] = wmma_bf16(a_cur, b_cur0, acc[0]);
    acc[1] = wmma_bf16(a_cur, b_cur1, acc[1]);
    acc[2] = wmma_bf16(a_cur, b_cur2, acc[2]);
    acc[3] = wmma_bf16(a_cur, b_cur3, acc[3]);
    a_cur = a_nxt;
    b_cur0 = b_nxt0; b_cur1 = b_nxt1; b_cur2 = b_nxt2; b_cur3 = b_nxt3;
  }
  acc[0] = wmma_bf16(a_cur, b_cur0, acc[0]);
  acc[1] = wmma_bf16(a_cur, b_cur1, acc[1]);
  acc[2] = wmma_bf16(a_cur, b_cur2, acc[2]);
  acc[3] = wmma_bf16(a_cur, b_cur3, acc[3]);
}

// --------------------------- prep kernels ----------------------------------

// w [R,C] fp32 -> wt [C,R] bf16 (transpose so WMMA B loads are contiguous)
__global__ void k_wprep(const float* __restrict__ w, unsigned short* __restrict__ wt,
                        int R, int C) {
  int i = blockIdx.x * 256 + threadIdx.x;
  if (i >= R * C) return;
  int r = i / C, c = i - r * C;
  wt[(size_t)c * R + r] = f2bf(w[i]);
}

// adaptive avg pool (torch windowing) from x [B,4096,512] into token buffer
__global__ void k_avgpool(const float* __restrict__ x, float* __restrict__ tokf,
                          unsigned short* __restrict__ tokb, int ohw, int segoff) {
  int i = blockIdx.x * 256 + threadIdx.x;
  int total = BATCH * ohw * ohw * CCH;
  if (i >= total) return;
  int c = i & (CCH - 1);
  int j = i >> 9;
  int q = j % ohw; j /= ohw;
  int p = j % ohw; int b = j / ohw;
  int hs = (p * 64) / ohw, he = ((p + 1) * 64 + ohw - 1) / ohw;
  int ws = (q * 64) / ohw, we = ((q + 1) * 64 + ohw - 1) / ohw;
  float sum = 0.f;
  for (int hh = hs; hh < he; ++hh)
    for (int ww = ws; ww < we; ++ww)
      sum += x[(((size_t)b * 4096) + hh * 64 + ww) * CCH + c];
  float mean = sum / (float)((he - hs) * (we - ws));
  size_t idx = ((size_t)b * NTOK + segoff + p * ohw + q) * CCH + c;
  tokf[idx] = mean;
  tokb[idx] = f2bf(mean);
}

// exact 4x4 max pool 64x64 -> 16x16
__global__ void k_maxpool(const float* __restrict__ x, float* __restrict__ tokf,
                          unsigned short* __restrict__ tokb) {
  int i = blockIdx.x * 256 + threadIdx.x;
  if (i >= BATCH * 256 * CCH) return;
  int c = i & (CCH - 1);
  int j = i >> 9;
  int q = j & 15; j >>= 4;
  int p = j & 15; int b = j >> 4;
  float mx = -3.0e38f;
  for (int hh = p * 4; hh < p * 4 + 4; ++hh)
    for (int ww = q * 4; ww < q * 4 + 4; ++ww) {
      float v = x[(((size_t)b * 4096) + hh * 64 + ww) * CCH + c];
      mx = fmaxf(mx, v);
    }
  size_t idx = ((size_t)b * NTOK + MXOFF + p * 16 + q) * CCH + c;
  tokf[idx] = mx;
  tokb[idx] = f2bf(mx);
}

// m rows into kv-input buffer (bf16) + zero the pad rows 2267..2271
__global__ void k_fill_m(const float* __restrict__ m, unsigned short* __restrict__ kvin) {
  size_t i = (size_t)blockIdx.x * 256 + threadIdx.x;
  size_t total = (size_t)BATCH * (NPAD - NTOK) * CCH;   // 8 * 1029 * 512
  if (i >= total) return;
  int c = (int)(i & (CCH - 1));
  size_t j = i >> 9;
  int t = (int)(j % (NPAD - NTOK));
  int b = (int)(j / (NPAD - NTOK));
  unsigned short v = 0;
  if (t < LTOK) v = f2bf(m[((size_t)b * LTOK + t) * CCH + c]);
  kvin[((size_t)b * NPAD + NTOK + t) * CCH + c] = v;
}

// --------------------------- WMMA GEMM kernels -----------------------------

// per-scale 1x1 conv residual: out = tok + tok @ W + bias (fp32 out, pre-LN)
__global__ void k_conv_gemm(const unsigned short* __restrict__ tokb,
                            const float* __restrict__ tokf, float* __restrict__ outp,
                            const unsigned short* __restrict__ Wt,
                            const float* __restrict__ bias, int segoff, int segM) {
  int lane = threadIdx.x;
  int mt = blockIdx.x, nt = blockIdx.y, b = blockIdx.z;
  size_t rowbase = (size_t)b * NTOK + segoff;
  v8f acc[4];
  gemm_strip4(tokb + rowbase * CCH, CCH, mt * 16, segM, Wt, CCH, nt * 64, CCH, lane, acc);
  int rb = (lane >> 4) * 8;
  for (int t = 0; t < 4; ++t) {
    int col = nt * 64 + t * 16 + (lane & 15);
    for (int r = 0; r < 8; ++r) {
      int row = mt * 16 + rb + r;
      if (row < segM) {
        size_t idx = (rowbase + row) * CCH + col;
        outp[idx] = tokf[idx] + acc[t][r] + bias[col];
      }
    }
  }
}

// mx conv result -> d_out second slot, [B,256,C] tokens -> [B,C,16,16]
__global__ void k_mx_out(const float* __restrict__ pconv, float* __restrict__ dout2) {
  int i = blockIdx.x * 256 + threadIdx.x;
  if (i >= BATCH * CCH * 256) return;
  int s = i & 255;
  int j = i >> 8;
  int c = j & (CCH - 1);
  int b = j >> 9;
  dout2[((size_t)b * CCH + c) * 256 + s] =
      pconv[((size_t)b * NTOK + MXOFF + s) * CCH + c];
}

// LayerNorm over C=512 of conv output -> bf16 rows [0,1243) of kvin
__global__ void k_layernorm(const float* __restrict__ pin, const float* __restrict__ g,
                            const float* __restrict__ be, unsigned short* __restrict__ kvin) {
  __shared__ float red[256];
  int row = blockIdx.x;                       // 0 .. B*NTOK-1
  int b = row / NTOK, t = row - b * NTOK;
  const float* pr = pin + (size_t)row * CCH;
  int tid = threadIdx.x;
  float a0 = pr[tid], a1 = pr[tid + 256];
  red[tid] = a0 + a1;
  __syncthreads();
  for (int o2 = 128; o2 > 0; o2 >>= 1) {
    if (tid < o2) red[tid] += red[tid + o2];
    __syncthreads();
  }
  float mu = red[0] * (1.0f / 512.0f);
  __syncthreads();
  float d0 = a0 - mu, d1 = a1 - mu;
  red[tid] = d0 * d0 + d1 * d1;
  __syncthreads();
  for (int o2 = 128; o2 > 0; o2 >>= 1) {
    if (tid < o2) red[tid] += red[tid + o2];
    __syncthreads();
  }
  float inv = rsqrtf(red[0] * (1.0f / 512.0f) + 1e-5f);
  unsigned short* orow = kvin + ((size_t)b * NPAD + t) * CCH;
  orow[tid]       = f2bf(d0 * inv * g[tid] + be[tid]);
  orow[tid + 256] = f2bf(d1 * inv * g[tid + 256] + be[tid + 256]);
}

// kv projection: kvin[B,NPAD,512] @ Wkv -> K [B,nh,NPAD,64], V^T [B,nh,64,NPAD]
__global__ void k_kv_gemm(const unsigned short* __restrict__ kvin,
                          const unsigned short* __restrict__ WkvT,
                          unsigned short* __restrict__ kbuf,
                          unsigned short* __restrict__ vtb) {
  int lane = threadIdx.x;
  int mt = blockIdx.x, nt = blockIdx.y, b = blockIdx.z;
  const unsigned short* A = kvin + (size_t)b * NPAD * CCH;
  v8f acc[4];
  gemm_strip4(A, CCH, mt * 16, NPAD, WkvT, CCH, nt * 64, 2 * CCH, lane, acc);
  int rb = (lane >> 4) * 8;
  for (int t = 0; t < 4; ++t) {
    int col = nt * 64 + t * 16 + (lane & 15);
    for (int r = 0; r < 8; ++r) {
      int n = mt * 16 + rb + r;
      unsigned short v = f2bf(acc[t][r]);
      if (col < CCH) {                                 // K half
        int h = col >> 6, dd = col & 63;
        kbuf[(((size_t)b * 8 + h) * NPAD + n) * 64 + dd] = v;
      } else {                                         // V half, transposed
        int c2 = col - CCH;
        int h = c2 >> 6, dd = c2 & 63;
        vtb[(((size_t)b * 8 + h) * 64 + dd) * NPAD + n] = v;
      }
    }
  }
}

// q projection (scale 1/sqrt(64) folded in): m rows of kvin -> q [B,nh,L,64]
__global__ void k_q_gemm(const unsigned short* __restrict__ kvin,
                         const unsigned short* __restrict__ WqT,
                         unsigned short* __restrict__ qb) {
  int lane = threadIdx.x;
  int mt = blockIdx.x, nt = blockIdx.y, b = blockIdx.z;
  const unsigned short* A = kvin + ((size_t)b * NPAD + NTOK) * CCH;
  v8f acc[4];
  gemm_strip4(A, CCH, mt * 16, LTOK, WqT, CCH, nt * 64, CCH, lane, acc);
  int rb = (lane >> 4) * 8;
  for (int t = 0; t < 4; ++t) {
    int col = nt * 64 + t * 16 + (lane & 15);
    int h = col >> 6, dd = col & 63;
    for (int r = 0; r < 8; ++r) {
      int l = mt * 16 + rb + r;
      qb[(((size_t)b * 8 + h) * LTOK + l) * 64 + dd] = f2bf(acc[t][r] * 0.125f);
    }
  }
}

// flash attention: one wave handles one 16-row Q tile of one (b,h)
__global__ void k_flash_attn(const unsigned short* __restrict__ qb,
                             const unsigned short* __restrict__ kbuf,
                             const unsigned short* __restrict__ vtb,
                             unsigned short* __restrict__ aob) {
  __shared__ float Sb[16][33];
  __shared__ float rm[16], rl[16], rc[16];
  int lane = threadIdx.x;
  int l0 = blockIdx.x * 16;
  int h = blockIdx.y, b = blockIdx.z;
  const unsigned short* Q = qb   + (((size_t)b * 8 + h) * LTOK) * 64;
  const unsigned short* K = kbuf + (((size_t)b * 8 + h) * NPAD) * 64;
  const unsigned short* V = vtb  + (((size_t)b * 8 + h) * 64) * NPAD;

  v16bf qa0 = load_frag(Q, 64, l0 + (lane & 15), 0, lane);
  v16bf qa1 = load_frag(Q, 64, l0 + (lane & 15), 32, lane);

  v8f zz = {0.f, 0.f, 0.f, 0.f, 0.f, 0.f, 0.f, 0.f};
  v8f o[4];
  for (int t = 0; t < 4; ++t) o[t] = zz;
  if (lane < 16) { rm[lane] = -3.0e38f; rl[lane] = 0.0f; }
  __syncthreads();

  for (int kc = 0; kc < NPAD / 32; ++kc) {
    int n0 = kc * 32;
    // speculative prefetch of next chunk (global_prefetch_b8)
    if (kc + 1 < NPAD / 32) {
      __builtin_prefetch(K + (size_t)(n0 + 32 + lane) * 64, 0, 0);       // 32 K rows
      __builtin_prefetch(V + (size_t)(lane * 2 + 0) * NPAD + n0 + 32, 0, 0);
      __builtin_prefetch(V + (size_t)(lane * 2 + 1) * NPAD + n0 + 32, 0, 0);
    }
    // one load cluster for all 4 K fragments of this 32-key chunk
    v16bf kb00 = load_frag(K, 64, n0 +      (lane & 15), 0,  lane);
    v16bf kb01 = load_frag(K, 64, n0 +      (lane & 15), 32, lane);
    v16bf kb10 = load_frag(K, 64, n0 + 16 + (lane & 15), 0,  lane);
    v16bf kb11 = load_frag(K, 64, n0 + 16 + (lane & 15), 32, lane);
    // issue V fragments early: softmax phase below hides their latency
    v16bf vb0 = load_frag(V, NPAD,      (lane & 15), n0, lane);
    v16bf vb1 = load_frag(V, NPAD, 16 + (lane & 15), n0, lane);
    v16bf vb2 = load_frag(V, NPAD, 32 + (lane & 15), n0, lane);
    v16bf vb3 = load_frag(V, NPAD, 48 + (lane & 15), n0, lane);
    // S = (q*scale) @ K^T : two 16x16 tiles
    v8f s0 = wmma_bf16(qa0, kb00, zz);
    s0 = wmma_bf16(qa1, kb01, s0);
    v8f s1 = wmma_bf16(qa0, kb10, zz);
    s1 = wmma_bf16(qa1, kb11, s1);
    {
      int col = lane & 15;
      int rb = (lane >> 4) * 8;
      for (int r = 0; r < 8; ++r) Sb[rb + r][col] = s0[r];
      for (int r = 0; r < 8; ++r) Sb[rb + r][16 + col] = s1[r];
    }
    __syncthreads();
    // online softmax stats per row (lanes 0..15, one row each)
    if (lane < 16) {
      int row = lane;
      float mo = rm[row];
      float mx = mo;
      for (int c = 0; c < 32; ++c) {
        float sv = (n0 + c) < NREAL ? Sb[row][c] : -3.0e38f;
        mx = fmaxf(mx, sv);
      }
      float corr = __expf(mo - mx);
      float ls = rl[row] * corr;
      for (int c = 0; c < 32; ++c) {
        float pv = ((n0 + c) < NREAL) ? __expf(Sb[row][c] - mx) : 0.0f;
        Sb[row][c] = pv;
        ls += pv;
      }
      rm[row] = mx; rl[row] = ls; rc[row] = corr;
    }
    __syncthreads();
    // rescale running output accumulators
    {
      int rb = (lane >> 4) * 8;
      for (int t = 0; t < 4; ++t)
        for (int r = 0; r < 8; ++r) o[t][r] *= rc[rb + r];
    }
    // repack probabilities into A-fragment layout (bf16)
    FragU pf;
    {
      int row = lane & 15, kh = (lane >> 4) & 1;
      for (int e = 0; e < 16; ++e) {
        int kk = ((e < 8) ? 0 : 16) + kh * 8 + (e & 7);
        pf.u[e] = f2bf(Sb[row][kk]);
      }
    }
    // O += P @ V_chunk  (V fragments were issued before softmax)
    o[0] = wmma_bf16(pf.bf, vb0, o[0]);
    o[1] = wmma_bf16(pf.bf, vb1, o[1]);
    o[2] = wmma_bf16(pf.bf, vb2, o[2]);
    o[3] = wmma_bf16(pf.bf, vb3, o[3]);
    __syncthreads();
  }
  // normalize and emit bf16 [B,L,C] for the output projection
  int rb = (lane >> 4) * 8;
  int col = lane & 15;
  for (int t = 0; t < 4; ++t)
    for (int r = 0; r < 8; ++r) {
      int row = rb + r;
      float val = o[t][r] / rl[row];
      aob[((size_t)b * LTOK + l0 + row) * CCH + h * 64 + t * 16 + col] = f2bf(val);
    }
}

// final projection: attn_out @ Wp + bp -> d_out (fp32)
__global__ void k_out_gemm(const unsigned short* __restrict__ aob,
                           const unsigned short* __restrict__ WpT,
                           const float* __restrict__ bp, float* __restrict__ dout) {
  int lane = threadIdx.x;
  int mt = blockIdx.x, nt = blockIdx.y;
  v8f acc[4];
  gemm_strip4(aob, CCH, mt * 16, BATCH * LTOK, WpT, CCH, nt * 64, CCH, lane, acc);
  int rb = (lane >> 4) * 8;
  for (int t = 0; t < 4; ++t) {
    int col = nt * 64 + t * 16 + (lane & 15);
    for (int r = 0; r < 8; ++r) {
      int row = mt * 16 + rb + r;
      dout[(size_t)row * CCH + col] = acc[t][r] + bp[col];
    }
  }
}

// --------------------------- host launcher ---------------------------------

extern "C" void kernel_launch(void* const* d_in, const int* in_sizes, int n_in,
                              void* d_out, int out_size, void* d_ws, size_t ws_size,
                              hipStream_t stream) {
  (void)in_sizes; (void)n_in; (void)out_size; (void)ws_size;
  const float* x    = (const float*)d_in[0];
  const float* m    = (const float*)d_in[1];
  const float* w_sc[5] = {(const float*)d_in[2], (const float*)d_in[4],
                          (const float*)d_in[6], (const float*)d_in[8],
                          (const float*)d_in[10]};
  const float* b_sc[5] = {(const float*)d_in[3], (const float*)d_in[5],
                          (const float*)d_in[7], (const float*)d_in[9],
                          (const float*)d_in[11]};
  const float* ln_g = (const float*)d_in[12];
  const float* ln_b = (const float*)d_in[13];
  const float* Wq   = (const float*)d_in[14];
  const float* Wkv  = (const float*)d_in[15];
  const float* Wp   = (const float*)d_in[16];
  const float* bp   = (const float*)d_in[17];

  char* ws = (char*)d_ws;
  size_t off = 0;
  auto alloc = [&](size_t bytes) {
    void* p = ws + off;
    off += (bytes + 255) & ~(size_t)255;
    return p;
  };
  float*          tokf = (float*)         alloc((size_t)BATCH * NTOK * CCH * 4);
  float*          pcv  = (float*)         alloc((size_t)BATCH * NTOK * CCH * 4);
  unsigned short* tokb = (unsigned short*)alloc((size_t)BATCH * NTOK * CCH * 2);
  unsigned short* kvin = (unsigned short*)alloc((size_t)BATCH * NPAD * CCH * 2);
  unsigned short* qb   = (unsigned short*)alloc((size_t)BATCH * 8 * LTOK * 64 * 2);
  unsigned short* kb   = (unsigned short*)alloc((size_t)BATCH * 8 * NPAD * 64 * 2);
  unsigned short* vtb  = (unsigned short*)alloc((size_t)BATCH * 8 * 64 * NPAD * 2);
  unsigned short* aob  = (unsigned short*)alloc((size_t)BATCH * LTOK * CCH * 2);
  unsigned short* WqT  = (unsigned short*)alloc((size_t)CCH * CCH * 2);
  unsigned short* WkvT = (unsigned short*)alloc((size_t)2 * CCH * CCH * 2);
  unsigned short* WpT  = (unsigned short*)alloc((size_t)CCH * CCH * 2);
  unsigned short* wsT[5];
  for (int i = 0; i < 5; ++i)
    wsT[i] = (unsigned short*)alloc((size_t)CCH * CCH * 2);

  // 1) weight prep (transpose + bf16)
  k_wprep<<<dim3((CCH * CCH + 255) / 256), 256, 0, stream>>>(Wq, WqT, CCH, CCH);
  k_wprep<<<dim3((CCH * 2 * CCH + 255) / 256), 256, 0, stream>>>(Wkv, WkvT, CCH, 2 * CCH);
  k_wprep<<<dim3((CCH * CCH + 255) / 256), 256, 0, stream>>>(Wp, WpT, CCH, CCH);
  for (int i = 0; i < 5; ++i)
    k_wprep<<<dim3((CCH * CCH + 255) / 256), 256, 0, stream>>>(w_sc[i], wsT[i], CCH, CCH);

  // 2) pools -> token buffer
  const int ohw[4]    = {21, 16, 13, 11};
  const int segoff[5] = {0, 441, 697, 866, MXOFF};
  const int seglen[5] = {441, 256, 169, 121, 256};
  for (int i = 0; i < 4; ++i) {
    int total = BATCH * ohw[i] * ohw[i] * CCH;
    k_avgpool<<<dim3((total + 255) / 256), 256, 0, stream>>>(x, tokf, tokb, ohw[i], segoff[i]);
  }
  k_maxpool<<<dim3((BATCH * 256 * CCH + 255) / 256), 256, 0, stream>>>(x, tokf, tokb);

  // 3) per-scale conv residual GEMMs (WMMA, 16x64 strips)
  for (int i = 0; i < 5; ++i)
    k_conv_gemm<<<dim3((seglen[i] + 15) / 16, CCH / 64, BATCH), 32, 0, stream>>>(
        tokb, tokf, pcv, wsT[i], b_sc[i], segoff[i], seglen[i]);

  // 4) mx output (pre-LN conv result), second slot of d_out
  float* dout  = (float*)d_out;
  float* dout2 = dout + (size_t)BATCH * LTOK * CCH;
  k_mx_out<<<dim3((BATCH * CCH * 256 + 255) / 256), 256, 0, stream>>>(pcv, dout2);

  // 5) LayerNorm -> kvin p-rows; m -> kvin m-rows (+ zero pad rows)
  k_layernorm<<<dim3(BATCH * NTOK), 256, 0, stream>>>(pcv, ln_g, ln_b, kvin);
  {
    size_t total = (size_t)BATCH * (NPAD - NTOK) * CCH;
    k_fill_m<<<dim3((unsigned)((total + 255) / 256)), 256, 0, stream>>>(m, kvin);
  }

  // 6) q / kv projections (WMMA, 16x64 strips)
  k_q_gemm<<<dim3(LTOK / 16, CCH / 64, BATCH), 32, 0, stream>>>(kvin, WqT, qb);
  k_kv_gemm<<<dim3(NPAD / 16, 2 * CCH / 64, BATCH), 32, 0, stream>>>(kvin, WkvT, kb, vtb);

  // 7) flash attention (WMMA QK^T + PV, online softmax)
  k_flash_attn<<<dim3(LTOK / 16, 8, BATCH), 32, 0, stream>>>(qb, kb, vtb, aob);

  // 8) output projection (WMMA, 16x64 strips) -> d_out
  k_out_gemm<<<dim3(BATCH * LTOK / 16, CCH / 64), 32, 0, stream>>>(aob, WpT, bp, dout);
}
